// GraphRespiratory_75788992905476
// MI455X (gfx1250) — matrix operand order repaired and checked by reading, not compiled, not run on hardware
//
#include <hip/hip_runtime.h>
#include <hip/hip_bf16.h>

typedef __attribute__((ext_vector_type(16))) _Float16 v16h;
typedef __attribute__((ext_vector_type(8)))  _Float16 v8h;
typedef __attribute__((ext_vector_type(8)))  float    v8f;

namespace {
constexpr int kN   = 50000;
constexpr int kE   = 800000;
constexpr int kET  = kN + kE;          // edges incl. self loops

// workspace layout (in floats)
constexpr size_t OFF_LOOP = 0;                        // 2N  loop_attr
constexpr size_t OFF_DEG  = OFF_LOOP + 2ull * kN;     // N   in-degree (dst)
constexpr size_t OFF_DEG2 = OFF_DEG  + (size_t)kN;    // N   src+dst degree
constexpr size_t OFF_S    = OFF_DEG2 + (size_t)kN;    // N   factor-graph accum
constexpr size_t OFF_AMAX = OFF_S    + (size_t)kN;    // 2N  (uint flipped max)
constexpr size_t OFF_DEN  = OFF_AMAX + 2ull * kN;     // 2N  softmax denom
constexpr size_t OFF_P    = OFF_DEN  + 2ull * kN;     // 16  folded edge-attn coeffs
constexpr size_t OFF_ASD  = OFF_P + 16;               // 4N  a_s/a_d per node
constexpr size_t OFF_EA   = OFF_ASD + 4ull * kN;      // 2*ET edge attention / alpha
constexpr size_t OFF_HB   = OFF_EA + 2ull * kET;      // 128N h (current layer)
constexpr size_t OFF_ACC  = OFF_HB + 128ull * kN;     // 128N message accumulator
constexpr size_t OFF_X2   = OFF_ACC + 128ull * kN;    // 64N  layer-2 input
constexpr size_t OFF_H2   = OFF_X2 + 64ull * kN;      // 64N  final node embedding
constexpr size_t OFF_PN   = OFF_H2 + 64ull * kN;      // 5N   p_n
constexpr size_t OFF_PE   = OFF_PN + 5ull * kN;       // 2E   p_e
constexpr size_t OFF_Q    = OFF_PE + 2ull * kE;       // 2N   h2 @ Wec
} // namespace

// ---- helpers -------------------------------------------------------------
__device__ __forceinline__ unsigned flipf(float f) {
  unsigned u = __float_as_uint(f);
  unsigned mask = (unsigned)(-(int)(u >> 31)) | 0x80000000u;
  return u ^ mask;
}
__device__ __forceinline__ float unflipf(unsigned u) {
  unsigned mask = ((u >> 31) - 1u) | 0x80000000u;
  return __uint_as_float(u ^ mask);
}

typedef __attribute__((address_space(3))) float lds_f32;

__global__ void kzero(float* p, long long n) {
  long long i = (long long)blockIdx.x * blockDim.x + threadIdx.x;
  long long st = (long long)gridDim.x * blockDim.x;
  for (; i < n; i += st) p[i] = 0.0f;
}

// degrees + per-dst edge-attr sums (for self-loop fill='mean')
__global__ void kdeg(const int* __restrict__ src, const int* __restrict__ dst,
                     const float* __restrict__ eattr,
                     float* deg, float* deg2, float* loop_attr) {
  int i = blockIdx.x * blockDim.x + threadIdx.x;
  if (i >= kE) return;
  int s = src[i], d = dst[i];
  atomicAdd(&deg[d], 1.0f);
  atomicAdd(&loop_attr[2 * d + 0], eattr[2 * i + 0]);
  atomicAdd(&loop_attr[2 * d + 1], eattr[2 * i + 1]);
  atomicAdd(&deg2[s], 1.0f);
  atomicAdd(&deg2[d], 1.0f);
}

__global__ void kloopdiv(const float* __restrict__ deg, float* loop_attr) {
  int n = blockIdx.x * blockDim.x + threadIdx.x;
  if (n >= kN) return;
  float dg = fmaxf(deg[n], 1.0f);
  loop_attr[2 * n + 0] /= dg;
  loop_attr[2 * n + 1] /= dg;
}

// P[k*2+h] = sum_c We[k,h*64+c] * att_e[h,c]   (folds e = eattr@We through att_e)
__global__ void kfoldP(const float* __restrict__ We, const float* __restrict__ att_e,
                       float* __restrict__ P) {
  int t = threadIdx.x;
  if (t >= 4) return;
  int k = t >> 1, h = t & 1;
  float s = 0.0f;
  for (int c = 0; c < 64; ++c) s += We[k * 128 + h * 64 + c] * att_e[h * 64 + c];
  P[k * 2 + h] = s;
}

// ---- WMMA GEMM: C[M x 128] = A[M x K] * B[K x 128], f32 in, f16 math, f32 acc
// Block = 256 threads = 8 waves; one 16-row M tile per block, 8 column tiles
// (one per wave). A panel (contiguous 16*K f32) arrives via async global->LDS
// DMA (ASYNCcnt); B is staged transposed as f16 so each lane's WMMA operands
// are two aligned ds_load_b128 per matrix per K-chunk. K unrolled -> K/32 WMMAs.
template <int K>
__global__ __launch_bounds__(256) void kgemm_wmma(const float* __restrict__ A,
                                                  const float* __restrict__ B,
                                                  float* __restrict__ C) {
  __shared__ float    sAraw[16 * K];
  __shared__ _Float16 sA[16 * K];
  __shared__ _Float16 sB[128 * K];
  const int t = threadIdx.x;
  const int row0 = blockIdx.x * 16;

  // --- async DMA of the contiguous A panel (16 rows x K f32 = 4*K b128 chunks)
  {
    lds_f32* as3 = (lds_f32*)sAraw;
    unsigned lbase = (unsigned)(size_t)as3;
    unsigned long long gbase = (unsigned long long)(A + (size_t)row0 * K);
    for (int j = t; j < 4 * K; j += 256) {
      unsigned loff = lbase + (unsigned)j * 16u;
      unsigned long long ga = gbase + (unsigned long long)j * 16u;
      asm volatile("global_load_async_to_lds_b128 %0, %1, off"
                   :: "v"(loff), "v"(ga) : "memory");
    }
    asm volatile("s_wait_asynccnt 0x0" ::: "memory");
  }
  // --- stage B transposed (sB[n*K+k]) as f16; global reads coalesced
  for (int j = t; j < 128 * K; j += 256) {
    int n = j & 127, k = j >> 7;
    sB[n * K + k] = (_Float16)B[j];
  }
  __syncthreads();
  // --- convert A panel f32 -> packed f16
  for (int j = t; j < 16 * K; j += 256) sA[j] = (_Float16)sAraw[j];
  __syncthreads();

  const int wave = t >> 5, lane = t & 31;
  const int m = lane & 15, g = lane >> 4;
  const int ncl = wave * 16 + m;          // B/C column for this lane
  v8f acc = {};
#pragma unroll
  for (int k0 = 0; k0 < K; k0 += 32) {
    // A 16x32 f16 layout (ISA 7.12.2): v0..3 -> K=8g+2v ; v4..7 -> K=16+8g+2v
    v8h alo = *(const v8h*)&sA[m * K + k0 + 8 * g];
    v8h ahi = *(const v8h*)&sA[m * K + k0 + 16 + 8 * g];
    // B 32x16 f16 layout: lane&15 = column, v0..3 -> K=8g+2v ; v4..7 -> K=16+8g+2v
    v8h blo = *(const v8h*)&sB[ncl * K + k0 + 8 * g];
    v8h bhi = *(const v8h*)&sB[ncl * K + k0 + 16 + 8 * g];
    v16h av = __builtin_shufflevector(alo, ahi, 0, 1, 2, 3, 4, 5, 6, 7,
                                      8, 9, 10, 11, 12, 13, 14, 15);
    v16h bv = __builtin_shufflevector(blo, bhi, 0, 1, 2, 3, 4, 5, 6, 7,
                                      8, 9, 10, 11, 12, 13, 14, 15);
    acc = __builtin_amdgcn_wmma_f32_16x16x32_f16(false, av, false, bv, (short)0,
                                                 acc, false, false);
  }
  // C layout: VGPR r -> row = r + 8g, col = lane&15
#pragma unroll
  for (int r = 0; r < 8; ++r)
    C[(size_t)(row0 + r + 8 * g) * 128 + ncl] = acc[r];
}

// per-node attention scalars a_s, a_d
__global__ void kasd(const float* __restrict__ hb, const float* __restrict__ att_s,
                     const float* __restrict__ att_d, float* __restrict__ asd) {
  int id = blockIdx.x * blockDim.x + threadIdx.x;
  if (id >= 2 * kN) return;
  int n = id >> 1, h = id & 1;
  float as = 0.0f, ad = 0.0f;
  const float* hp = hb + (size_t)n * 128 + h * 64;
  for (int c = 0; c < 64; ++c) {
    float hv = hp[c];
    as += hv * att_s[h * 64 + c];
    ad += hv * att_d[h * 64 + c];
  }
  asd[n * 4 + h] = as;
  asd[n * 4 + 2 + h] = ad;
}

// edge attention score + segment max (flipped-uint atomicMax)
__global__ void kedge_a(const int* __restrict__ src, const int* __restrict__ dst,
                        const float* __restrict__ eattr, const float* __restrict__ loop_attr,
                        const float* __restrict__ P, const float* __restrict__ asd,
                        float* __restrict__ edge_a, unsigned* __restrict__ amax) {
  int i = blockIdx.x * blockDim.x + threadIdx.x;
  if (i >= kET) return;
  int s, d; float e0, e1;
  if (i < kE) { s = src[i]; d = dst[i]; e0 = eattr[2 * i]; e1 = eattr[2 * i + 1]; }
  else { s = d = i - kE; e0 = loop_attr[2 * s]; e1 = loop_attr[2 * s + 1]; }
#pragma unroll
  for (int h = 0; h < 2; ++h) {
    float a = asd[s * 4 + h] + asd[d * 4 + 2 + h] + e0 * P[h] + e1 * P[2 + h];
    a = (a > 0.0f) ? a : 0.2f * a;                 // leaky_relu(0.2)
    edge_a[(size_t)i * 2 + h] = a;
    atomicMax(&amax[d * 2 + h], flipf(a));
  }
}

__global__ void kedge_exp(const int* __restrict__ dst, const unsigned* __restrict__ amax,
                          float* __restrict__ edge_a, float* __restrict__ den) {
  int i = blockIdx.x * blockDim.x + threadIdx.x;
  if (i >= kET) return;
  int d = (i < kE) ? dst[i] : (i - kE);
#pragma unroll
  for (int h = 0; h < 2; ++h) {
    float m = unflipf(amax[d * 2 + h]);
    float ea = __expf(edge_a[(size_t)i * 2 + h] - m);
    edge_a[(size_t)i * 2 + h] = ea;
    atomicAdd(&den[d * 2 + h], ea);
  }
}

__global__ void kedge_alpha(const int* __restrict__ dst, const float* __restrict__ den,
                            float* __restrict__ edge_a) {
  int i = blockIdx.x * blockDim.x + threadIdx.x;
  if (i >= kET) return;
  int d = (i < kE) ? dst[i] : (i - kE);
#pragma unroll
  for (int h = 0; h < 2; ++h)
    edge_a[(size_t)i * 2 + h] /= (den[d * 2 + h] + 1e-16f);
}

// message passing: acc[d, h, c] += alpha[e,h] * h[s, h, c]  (thread per (edge, c))
__global__ void kedge_msg(const int* __restrict__ src, const int* __restrict__ dst,
                          const float* __restrict__ edge_a, const float* __restrict__ hb,
                          float* __restrict__ acc) {
  long long t = (long long)blockIdx.x * blockDim.x + threadIdx.x;
  if (t >= (long long)kET * 64) return;
  int i = (int)(t >> 6), c = (int)(t & 63);
  int s, d;
  if (i < kE) { s = src[i]; d = dst[i]; } else { s = d = i - kE; }
  const float* hs = hb + (size_t)s * 128;
  __builtin_prefetch(hs + c, 0, 0);                // global_prefetch_b8 (gather row)
  float a0 = edge_a[(size_t)i * 2 + 0];
  float a1 = edge_a[(size_t)i * 2 + 1];
  atomicAdd(&acc[(size_t)d * 128 + c],       a0 * hs[c]);
  atomicAdd(&acc[(size_t)d * 128 + 64 + c],  a1 * hs[64 + c]);
}

// head mean + bias (+ optional relu)
__global__ void khead(const float* __restrict__ acc, const float* __restrict__ bias,
                      float* __restrict__ out, int do_relu) {
  long long t = (long long)blockIdx.x * blockDim.x + threadIdx.x;
  if (t >= (long long)kN * 64) return;
  int n = (int)(t >> 6), c = (int)(t & 63);
  float v = 0.5f * (acc[(size_t)n * 128 + c] + acc[(size_t)n * 128 + 64 + c]) + bias[c];
  if (do_relu) v = fmaxf(v, 0.0f);
  out[(size_t)n * 64 + c] = v;
}

// node logits -> softmax p_n ; also q = h2 @ Wec
__global__ void knodehead(const float* __restrict__ h2, const float* __restrict__ Wn,
                          const float* __restrict__ bn, const float* __restrict__ Wec,
                          float* __restrict__ p_n, float* __restrict__ q) {
  int n = blockIdx.x * blockDim.x + threadIdx.x;
  if (n >= kN) return;
  float z[5], qq[2] = {0.0f, 0.0f};
#pragma unroll
  for (int j = 0; j < 5; ++j) z[j] = bn[j];
  const float* hp = h2 + (size_t)n * 64;
  for (int c = 0; c < 64; ++c) {
    float hv = hp[c];
#pragma unroll
    for (int j = 0; j < 5; ++j) z[j] += hv * Wn[c * 5 + j];
    qq[0] += hv * Wec[c * 2 + 0];
    qq[1] += hv * Wec[c * 2 + 1];
  }
  float m = z[0];
#pragma unroll
  for (int j = 1; j < 5; ++j) m = fmaxf(m, z[j]);
  float sum = 0.0f;
#pragma unroll
  for (int j = 0; j < 5; ++j) { z[j] = __expf(z[j] - m); sum += z[j]; }
  float inv = 1.0f / sum;
#pragma unroll
  for (int j = 0; j < 5; ++j) p_n[(size_t)n * 5 + j] = z[j] * inv;
  q[n * 2 + 0] = qq[0];
  q[n * 2 + 1] = qq[1];
}

__global__ void kedgehead(const int* __restrict__ src, const int* __restrict__ dst,
                          const float* __restrict__ q, const float* __restrict__ bec,
                          float* __restrict__ p_e) {
  int i = blockIdx.x * blockDim.x + threadIdx.x;
  if (i >= kE) return;
  int s = src[i], d = dst[i];
  float l0 = 0.5f * (q[s * 2 + 0] + q[d * 2 + 0]) + bec[0];
  float l1 = 0.5f * (q[s * 2 + 1] + q[d * 2 + 1]) + bec[1];
  float m = fmaxf(l0, l1);
  float e0 = __expf(l0 - m), e1 = __expf(l1 - m);
  float inv = 1.0f / (e0 + e1);
  p_e[(size_t)i * 2 + 0] = e0 * inv;
  p_e[(size_t)i * 2 + 1] = e1 * inv;
}

// factor-graph edge step (GAMMA=1): update p_e, accumulate eab into s[src],s[dst]
__global__ void kfact_e(const int* __restrict__ src, const int* __restrict__ dst,
                        const float* __restrict__ p_n, const float* __restrict__ edge_fw,
                        float* __restrict__ p_e, float* __restrict__ sacc) {
  int i = blockIdx.x * blockDim.x + threadIdx.x;
  if (i >= kE) return;
  float avg_ef = 0.25f * (edge_fw[6] + edge_fw[7] + edge_fw[8] + edge_fw[9]);
  int s = src[i], d = dst[i];
  float ena = fmaxf(1.0f - p_n[(size_t)s * 5], 1.0f - p_n[(size_t)d * 5]);
  float p0 = p_e[(size_t)i * 2 + 0];
  float p1 = p_e[(size_t)i * 2 + 1] * (1.0f + ena * avg_ef);
  float inv = 1.0f / (p0 + p1);
  p0 *= inv; p1 *= inv;
  p_e[(size_t)i * 2 + 0] = p0;
  p_e[(size_t)i * 2 + 1] = p1;
  float eab = 1.0f - p0;
  atomicAdd(&sacc[s], eab);
  atomicAdd(&sacc[d], eab);
}

// factor-graph node step; consumes and re-zeros s[]
__global__ void kfact_n(const float* __restrict__ deg2, const float* __restrict__ node_fw,
                        float* __restrict__ p_n, float* __restrict__ sacc) {
  int n = blockIdx.x * blockDim.x + threadIdx.x;
  if (n >= kN) return;
  float avg_nf = 0.25f * (node_fw[3] + node_fw[5] + node_fw[7] + node_fw[9]);
  float mean = sacc[n] / (deg2[n] + 1e-6f);
  float f = 1.0f + mean * avg_nf;
  float p[5];
  p[0] = p_n[(size_t)n * 5];
  float sum = p[0];
#pragma unroll
  for (int j = 1; j < 5; ++j) { p[j] = p_n[(size_t)n * 5 + j] * f; sum += p[j]; }
  float inv = 1.0f / sum;
#pragma unroll
  for (int j = 0; j < 5; ++j) p_n[(size_t)n * 5 + j] = p[j] * inv;
  sacc[n] = 0.0f;
}

__global__ void kout_node(const float* __restrict__ p_n, float* __restrict__ out) {
  long long t = (long long)blockIdx.x * blockDim.x + threadIdx.x;
  if (t >= (long long)kN * 5) return;
  out[t] = logf(p_n[t] + 1e-9f);
}
__global__ void kout_edge(const float* __restrict__ p_e, float* __restrict__ out) {
  long long t = (long long)blockIdx.x * blockDim.x + threadIdx.x;
  if (t >= (long long)kE * 2) return;
  out[(long long)kN * 5 + t] = logf(p_e[t] + 1e-9f);
}

// --------------------------------------------------------------------------
extern "C" void kernel_launch(void* const* d_in, const int* in_sizes, int n_in,
                              void* d_out, int out_size, void* d_ws, size_t ws_size,
                              hipStream_t stream) {
  const float* x      = (const float*)d_in[0];
  const int*   ei     = (const int*)d_in[1];
  const float* eattr  = (const float*)d_in[2];
  const float* W1     = (const float*)d_in[3];
  const float* att_s1 = (const float*)d_in[4];
  const float* att_d1 = (const float*)d_in[5];
  const float* We1    = (const float*)d_in[6];
  const float* att_e1 = (const float*)d_in[7];
  const float* b1     = (const float*)d_in[8];
  const float* W2     = (const float*)d_in[9];
  const float* att_s2 = (const float*)d_in[10];
  const float* att_d2 = (const float*)d_in[11];
  const float* We2    = (const float*)d_in[12];
  const float* att_e2 = (const float*)d_in[13];
  const float* b2     = (const float*)d_in[14];
  const float* Wn     = (const float*)d_in[15];
  const float* bn     = (const float*)d_in[16];
  const float* Wec    = (const float*)d_in[17];
  const float* bec    = (const float*)d_in[18];
  const float* node_fw = (const float*)d_in[19];
  const float* edge_fw = (const float*)d_in[20];

  const int* src = ei;
  const int* dst = ei + kE;

  float* ws = (float*)d_ws;
  float*    loop_attr = ws + OFF_LOOP;
  float*    deg   = ws + OFF_DEG;
  float*    deg2  = ws + OFF_DEG2;
  float*    sacc  = ws + OFF_S;
  unsigned* amax  = (unsigned*)(ws + OFF_AMAX);
  float*    den   = ws + OFF_DEN;
  float*    P     = ws + OFF_P;
  float*    asd   = ws + OFF_ASD;
  float*    ea    = ws + OFF_EA;
  float*    hb    = ws + OFF_HB;
  float*    acc   = ws + OFF_ACC;
  float*    x2    = ws + OFF_X2;
  float*    h2    = ws + OFF_H2;
  float*    p_n   = ws + OFF_PN;
  float*    p_e   = ws + OFF_PE;
  float*    q     = ws + OFF_Q;

  const int B = 256;
  auto blocks = [](long long n) { return (unsigned)((n + 255) / 256); };

  // ---- init ----
  kzero<<<4096, B, 0, stream>>>(ws + OFF_LOOP, 5ll * kN);           // loop_attr,deg,deg2,s
  kzero<<<4096, B, 0, stream>>>(ws + OFF_AMAX, 4ll * kN);           // amax (0 == flipped min), den
  kzero<<<8192, B, 0, stream>>>(acc, 128ll * kN);
  kdeg<<<blocks(kE), B, 0, stream>>>(src, dst, eattr, deg, deg2, loop_attr);
  kloopdiv<<<blocks(kN), B, 0, stream>>>(deg, loop_attr);

  // ---- GAT layer 1 (input x[N,128], W1[128,128]) ----
  kfoldP<<<1, 64, 0, stream>>>(We1, att_e1, P);
  kgemm_wmma<128><<<kN / 16, 256, 0, stream>>>(x, W1, hb);
  kasd<<<blocks(2ll * kN), B, 0, stream>>>(hb, att_s1, att_d1, asd);
  kedge_a<<<blocks(kET), B, 0, stream>>>(src, dst, eattr, loop_attr, P, asd, ea, amax);
  kedge_exp<<<blocks(kET), B, 0, stream>>>(dst, amax, ea, den);
  kedge_alpha<<<blocks(kET), B, 0, stream>>>(dst, den, ea);
  kedge_msg<<<blocks((long long)kET * 64), B, 0, stream>>>(src, dst, ea, hb, acc);
  khead<<<blocks((long long)kN * 64), B, 0, stream>>>(acc, b1, x2, /*relu=*/1);

  // ---- GAT layer 2 (input x2[N,64], W2[64,128]) ----
  kzero<<<4096, B, 0, stream>>>(ws + OFF_AMAX, 4ll * kN);
  kzero<<<8192, B, 0, stream>>>(acc, 128ll * kN);
  kfoldP<<<1, 64, 0, stream>>>(We2, att_e2, P);
  kgemm_wmma<64><<<kN / 16, 256, 0, stream>>>(x2, W2, hb);
  kasd<<<blocks(2ll * kN), B, 0, stream>>>(hb, att_s2, att_d2, asd);
  kedge_a<<<blocks(kET), B, 0, stream>>>(src, dst, eattr, loop_attr, P, asd, ea, amax);
  kedge_exp<<<blocks(kET), B, 0, stream>>>(dst, amax, ea, den);
  kedge_alpha<<<blocks(kET), B, 0, stream>>>(dst, den, ea);
  kedge_msg<<<blocks((long long)kET * 64), B, 0, stream>>>(src, dst, ea, hb, acc);
  khead<<<blocks((long long)kN * 64), B, 0, stream>>>(acc, b2, h2, /*relu=*/0);

  // ---- heads + factor graph ----
  knodehead<<<blocks(kN), B, 0, stream>>>(h2, Wn, bn, Wec, p_n, q);
  kedgehead<<<blocks(kE), B, 0, stream>>>(src, dst, q, bec, p_e);
  for (int it = 0; it < 2; ++it) {
    kfact_e<<<blocks(kE), B, 0, stream>>>(src, dst, p_n, edge_fw, p_e, sacc);
    kfact_n<<<blocks(kN), B, 0, stream>>>(deg2, node_fw, p_n, sacc);  // re-zeros sacc
  }
  kout_node<<<blocks(5ll * kN), B, 0, stream>>>(p_n, (float*)d_out);
  kout_edge<<<blocks(2ll * kE), B, 0, stream>>>(p_e, (float*)d_out);

  (void)in_sizes; (void)n_in; (void)out_size; (void)ws_size;
}
